// RelationalGraphConvolution_85796266705313
// MI455X (gfx1250) — compile-verified, matrix operand order checked by reading, not run on hardware
//
#include <hip/hip_runtime.h>

// ---------------------------------------------------------------------------
// RGCN basis-decomposition forward for MI455X (gfx1250, wave32, WMMA).
//
// out[4096,128] = adj[4096, 8*4096] @ (features @ (comp·bases))  + bias
//
// Roofline: adj = 512 MB fp32 (one-shot stream, ~22 us @ 23.3 TB/s),
// 34.4 GFLOP => 67 FLOP/B => must use WMMA bf16 (f32 accumulate) to stay
// HBM-bound. fwT (8 MB bf16) lives in the 192 MB L2 and is re-read per
// workgroup (~0.5 GB L2 traffic with 4x A-reuse per B fragment).
// ---------------------------------------------------------------------------

typedef __attribute__((ext_vector_type(16))) __bf16 bf16x16;
typedef __attribute__((ext_vector_type(8)))  __bf16 bf16x8;
typedef __attribute__((ext_vector_type(4)))  __bf16 bf16x4;
typedef __attribute__((ext_vector_type(8)))  float  f32x8;
typedef __attribute__((ext_vector_type(4)))  float  f32x4;

#define N_NODES 4096
#define RELS    8
#define IN_F    128
#define OUT_F   128
#define BASES   8
#define KTOT    (RELS * N_NODES)        // 32768

#define SPLITK  16
#define KCHUNK  (KTOT / SPLITK)         // 2048
#define STAGE_K 64
#define NSTAGES (KCHUNK / STAGE_K)      // 32
#define ROWS    128                      // M rows per workgroup
#define LDSP    72                       // padded LDS row stride (bf16 units)

union Frag { bf16x16 v; bf16x8 h[2]; };

// Assemble a 16x32 (A) or 32x16 (B) bf16 operand fragment from two 16-byte
// loads, matching the CDNA5 16-bit operand VGPR layout:
//   lane L: row/col = L%16, VGPR0-3 hold K = ksel..ksel+7,
//           VGPR4-7 hold K = 16+ksel..16+ksel+7, ksel = (L/16)*8.
__device__ __forceinline__ bf16x16 make_frag(const __bf16* p) {
  Frag f;
  f.h[0] = *(const bf16x8*)(p);
  f.h[1] = *(const bf16x8*)(p + 16);
  return f.v;
}

// ---------------------------------------------------------------------------
// Kernel 1: wT[r][o][i] = sum_b comp[r,b]*bases[b,i,o]  (transposed, bf16)
//           featbf = bf16(features);   out[n][o] = bias[o]
// ---------------------------------------------------------------------------
__global__ __launch_bounds__(256)
void rgcn_prep(const float* __restrict__ features,
               const float* __restrict__ bases,
               const float* __restrict__ comp,
               const float* __restrict__ bias,
               __bf16* __restrict__ wT,
               __bf16* __restrict__ featbf,
               float*  __restrict__ out) {
  const int i = blockIdx.x * 256 + threadIdx.x;
  if (i < RELS * IN_F * OUT_F) {
    const int r  = i >> 14;
    const int ii = (i >> 7) & (IN_F - 1);
    const int oo = i & (OUT_F - 1);
    float acc = 0.f;
#pragma unroll
    for (int b = 0; b < BASES; ++b)
      acc += comp[r * BASES + b] * bases[b * (IN_F * OUT_F) + ii * OUT_F + oo];
    wT[r * (IN_F * OUT_F) + oo * IN_F + ii] = (__bf16)acc;  // transposed store
  }
  if (i < N_NODES * IN_F)  featbf[i] = (__bf16)features[i];
  if (i < N_NODES * OUT_F) out[i] = bias[i & (OUT_F - 1)];
}

// ---------------------------------------------------------------------------
// Kernel 2: fw = features @ weights[r], stored transposed:
//           fwT[o][r*N + n] = fw[r][n][o]   (bf16, 8 MB -> L2 resident)
// One wave per 16x16 C tile; K = 128 (4 WMMAs).
// ---------------------------------------------------------------------------
__global__ __launch_bounds__(256)
void rgcn_fw(const __bf16* __restrict__ featbf,
             const __bf16* __restrict__ wT,
             __bf16* __restrict__ fwT) {
  const int lane = threadIdx.x & 31;
  const int wid  = threadIdx.x >> 5;
  const int flat = blockIdx.x * 8 + wid;      // 16384 waves total
  const int r    = flat >> 11;                // 8 relations
  const int mb   = (flat >> 3) & 255;         // 256 node tiles
  const int nb   = flat & 7;                  // 8 out-feature tiles
  const int l15  = lane & 15;
  const int ksel = (lane >> 4) * 8;

  const __bf16* Abase = featbf + (size_t)(mb * 16 + l15) * IN_F + ksel;
  const __bf16* Bbase = wT + (size_t)r * (IN_F * OUT_F)
                           + (size_t)(nb * 16 + l15) * IN_F + ksel;

  f32x8 acc = {};
#pragma unroll
  for (int kk = 0; kk < IN_F / 32; ++kk) {
    bf16x16 a = make_frag(Abase + kk * 32);
    bf16x16 b = make_frag(Bbase + kk * 32);
    acc = __builtin_amdgcn_wmma_f32_16x16x32_bf16(false, a, false, b,
                                                  (short)0, acc, false, false);
  }
  // C layout: VGPR v, lane L -> M = v + 8*(L/16), N = L%16
  const int node0 = mb * 16 + 8 * (lane >> 4);
  const int o     = nb * 16 + l15;
#pragma unroll
  for (int v = 0; v < 8; ++v)
    fwT[(size_t)o * KTOT + (size_t)r * N_NODES + node0 + v] = (__bf16)acc[v];
}

// ---------------------------------------------------------------------------
// Kernel 3: out += adj @ fwT^T  (split-K, atomic f32 accumulate)
// Workgroup: 128 rows x 128 cols x KCHUNK. 8 waves = 2 m-groups x 4 n-groups;
// each wave: MT=4 m-tiles x NT=2 n-tiles (4x2 f32x8 accumulators).
// adj staged fp32->bf16 into double-buffered LDS (padded stride 72).
// ---------------------------------------------------------------------------
__global__ __launch_bounds__(256)
void rgcn_main(const float* __restrict__ adj,
               const __bf16* __restrict__ fwT,
               float* __restrict__ out) {
  __shared__ __bf16 tile[2][ROWS * LDSP];    // 2 x 18 KB

  const int tid  = threadIdx.x;
  const int lane = tid & 31;
  const int wid  = tid >> 5;
  const int l15  = lane & 15;
  const int ksel = (lane >> 4) * 8;
  const int mgrp = wid & 1;                  // 2 groups of 64 rows
  const int ngrp = wid >> 1;                 // 4 groups of 32 cols
  const size_t mbase  = (size_t)blockIdx.x * ROWS;
  const int    kbase0 = blockIdx.y * KCHUNK;

  const __bf16* B0 = fwT + (size_t)(ngrp * 32 + l15) * KTOT + kbase0 + ksel;
  const __bf16* B1 = B0 + (size_t)16 * KTOT;

  f32x8 acc[4][2] = {};

  // Stage prefetch registers: 32 floats/thread (128 rows x 64 K per stage).
  f32x4 pf[8];
#pragma unroll
  for (int i = 0; i < 8; ++i) {
    const int f = tid * 4 + i * 1024;
    pf[i] = *(const f32x4*)(adj + (mbase + (f >> 6)) * (size_t)KTOT
                                + kbase0 + (f & 63));
  }

  for (int s = 0; s < NSTAGES; ++s) {
    __bf16* buf = tile[s & 1];
    // Convert fp32 -> bf16 and store stage into LDS.
#pragma unroll
    for (int i = 0; i < 8; ++i) {
      const int f = tid * 4 + i * 1024;
      bf16x4 q;
      q.x = (__bf16)pf[i].x; q.y = (__bf16)pf[i].y;
      q.z = (__bf16)pf[i].z; q.w = (__bf16)pf[i].w;
      *(bf16x4*)(buf + (f >> 6) * LDSP + (f & 63)) = q;
    }
    // Prefetch next stage from HBM (overlaps with this stage's WMMAs).
    if (s + 1 < NSTAGES) {
      const int kb = kbase0 + (s + 1) * STAGE_K;
#pragma unroll
      for (int i = 0; i < 8; ++i) {
        const int f = tid * 4 + i * 1024;
        pf[i] = *(const f32x4*)(adj + (mbase + (f >> 6)) * (size_t)KTOT
                                    + kb + (f & 63));
      }
    }
    // Single barrier per stage: __syncthreads waits dscnt==0, so all waves'
    // prior-stage ds_loads have landed in VGPRs before any buffer reuse.
    __syncthreads();

    const int ks = s * STAGE_K;
#pragma unroll
    for (int kk = 0; kk < STAGE_K / 32; ++kk) {
      bf16x16 b0 = make_frag(B0 + ks + kk * 32);
      bf16x16 b1 = make_frag(B1 + ks + kk * 32);
#pragma unroll
      for (int mt = 0; mt < 4; ++mt) {
        const __bf16* ap = buf + (mgrp * 64 + mt * 16 + l15) * LDSP
                               + kk * 32 + ksel;
        bf16x16 a = make_frag(ap);
        acc[mt][0] = __builtin_amdgcn_wmma_f32_16x16x32_bf16(
            false, a, false, b0, (short)0, acc[mt][0], false, false);
        acc[mt][1] = __builtin_amdgcn_wmma_f32_16x16x32_bf16(
            false, a, false, b1, (short)0, acc[mt][1], false, false);
      }
    }
  }

  // Split-K accumulate into out (hardware f32 atomics).
#pragma unroll
  for (int mt = 0; mt < 4; ++mt) {
    const size_t row = mbase + mgrp * 64 + mt * 16 + 8 * (lane >> 4);
#pragma unroll
    for (int nt = 0; nt < 2; ++nt) {
      const int col = ngrp * 32 + nt * 16 + l15;
      float* op = out + row * OUT_F + col;
#pragma unroll
      for (int v = 0; v < 8; ++v)
        unsafeAtomicAdd(op + (size_t)v * OUT_F, acc[mt][nt][v]);
    }
  }
}

// ---------------------------------------------------------------------------
// Host launcher
// ---------------------------------------------------------------------------
extern "C" void kernel_launch(void* const* d_in, const int* in_sizes, int n_in,
                              void* d_out, int out_size, void* d_ws, size_t ws_size,
                              hipStream_t stream) {
  (void)in_sizes; (void)n_in; (void)out_size; (void)ws_size;
  const float* adj      = (const float*)d_in[0];
  const float* features = (const float*)d_in[1];
  const float* bases    = (const float*)d_in[2];
  const float* comp     = (const float*)d_in[3];
  const float* bias     = (const float*)d_in[4];
  float* out = (float*)d_out;

  char* ws = (char*)d_ws;
  __bf16* fwT    = (__bf16*)(ws);                                   // 8 MB
  __bf16* featbf = (__bf16*)(ws + (size_t)OUT_F * KTOT * 2);        // 1 MB
  __bf16* wT     = (__bf16*)(ws + (size_t)OUT_F * KTOT * 2
                                + (size_t)N_NODES * IN_F * 2);      // 256 KB

  rgcn_prep<<<dim3((N_NODES * IN_F + 255) / 256), 256, 0, stream>>>(
      features, bases, comp, bias, wT, featbf, out);
  rgcn_fw<<<dim3((RELS * 256 * 8) / 8), 256, 0, stream>>>(featbf, wT, fwT);
  rgcn_main<<<dim3(N_NODES / ROWS, SPLITK), 256, 0, stream>>>(adj, fwT, out);
}
